// SACRSN_v50_23536420782643
// MI455X (gfx1250) — compile-verified, matrix operand order checked by reading, not run on hardware
//
#include <hip/hip_runtime.h>
#include <math.h>

typedef __bf16 bf16;
typedef __attribute__((ext_vector_type(16))) __bf16 v16bf;
typedef __attribute__((ext_vector_type(8)))  __bf16 v8bf;
typedef __attribute__((ext_vector_type(8)))  float  v8f;

#define B_ 32
#define T_ 32
#define D_ 256
#define S_ 32
#define C_ 128
#define V_ 32000
#define R_ 8
#define D2_ 512
#define PONDER_ 0.01f
#define TWO_PI_ 6.28318530717958647692f

// ---------------- workspace layout (float slots) ----------------
constexpr size_t NBTD   = (size_t)B_ * T_ * D_;          // 262144
constexpr size_t OFF_ER    = 0;
constexpr size_t OFF_EI    = OFF_ER + NBTD;
constexpr size_t OFF_GWR   = OFF_EI + NBTD;
constexpr size_t OFF_GWI   = OFF_GWR + (size_t)B_ * D_;
constexpr size_t OFF_MEMR  = OFF_GWI + (size_t)B_ * D_;
constexpr size_t OFF_MEMI  = OFF_MEMR + (size_t)B_ * S_ * D_;
constexpr size_t OFF_PANG  = OFF_MEMI + (size_t)B_ * S_ * D_;
constexpr size_t OFF_ACT   = OFF_PANG + (size_t)B_ * D_;
constexpr size_t OFF_STOP  = OFF_ACT + B_;
constexpr size_t OFF_CSR   = OFF_STOP + B_;
constexpr size_t OFF_CSI   = OFF_CSR + (size_t)B_ * D_;
constexpr size_t OFF_FLAT  = OFF_CSI + (size_t)B_ * D_;
constexpr size_t OFF_ZQ    = OFF_FLAT + (size_t)B_ * D2_;
constexpr size_t OFF_QKV   = OFF_ZQ + (size_t)B_ * D2_;
constexpr size_t OFF_BIAS  = OFF_QKV + (size_t)6 * B_ * D_;
constexpr size_t OFF_EMBSQ = OFF_BIAS + 6 * D_;
constexpr size_t OFF_AMLOG = OFF_EMBSQ + C_;
constexpr size_t OFF_VQLOG = OFF_AMLOG + (size_t)T_ * R_ * B_;
constexpr size_t OFF_PHLOG = OFF_VQLOG + (size_t)T_ * R_ * B_;
constexpr size_t OFF_IDXLOG= OFF_PHLOG + (size_t)T_ * R_ * B_;
constexpr size_t OFF_OUTSEQ= OFF_IDXLOG + (size_t)T_ * R_ * B_;
constexpr size_t OFF_WCAT  = OFF_OUTSEQ + (size_t)B_ * T_ * D2_;        // bf16 [6][256][512]
constexpr size_t OFF_DECWBF= OFF_WCAT + (size_t)6 * D2_ * D_ / 2;       // bf16 [32000][512]
constexpr size_t OFF_OUTBF = OFF_DECWBF + (size_t)V_ * D2_ / 2;         // bf16 [1024][512]
constexpr size_t ZERO_N    = OFF_CSR - OFF_GWR;                         // gw/mem/pang/act/stop

// ---------------- output layout (element slots) ----------------
constexpr size_t OUT_LOGITS = 0;
constexpr size_t OUT_STATS  = (size_t)B_ * T_ * V_;
constexpr size_t OUT_MEMR   = OUT_STATS + 5;
constexpr size_t OUT_MEMI   = OUT_MEMR + (size_t)B_ * S_ * D_;
constexpr size_t OUT_IDX    = OUT_MEMI + (size_t)B_ * S_ * D_;

// ------------------- helpers -------------------
// 16-bit WMMA operand fragment for one 32-wide K chunk: per lane the 16
// elements are two contiguous runs of 8 (k in [8h,8h+8) and [16+8h,16+8h+8)),
// i.e. two 16-byte (b128) loads from a k-contiguous row.
__device__ __forceinline__ v16bf load_frag(const bf16* p, int h) {
  v8bf lo = *(const v8bf*)(p + 8 * h);
  v8bf hi = *(const v8bf*)(p + 16 + 8 * h);
  return __builtin_shufflevector(lo, hi, 0, 1, 2, 3, 4, 5, 6, 7,
                                 8, 9, 10, 11, 12, 13, 14, 15);
}

__device__ __forceinline__ float block_reduce_sum(float v, float* sm) {
  int tid = threadIdx.x;
  sm[tid] = v;
  __syncthreads();
  for (int s = 128; s > 0; s >>= 1) {
    if (tid < s) sm[tid] += sm[tid + s];
    __syncthreads();
  }
  float r = sm[0];
  __syncthreads();
  return r;
}

__device__ __forceinline__ float sigmoidf_(float x) { return 1.f / (1.f + expf(-x)); }

// ------------------- setup kernels -------------------
__global__ void zero_state_kernel(float* ws) {
  size_t i = (size_t)blockIdx.x * blockDim.x + threadIdx.x;
  if (i < ZERO_N) ws[OFF_GWR + i] = 0.f;
}

__global__ void embed_kernel(const int* __restrict__ xseq, const float* __restrict__ encw,
                             float* __restrict__ ws) {
  size_t i = (size_t)blockIdx.x * blockDim.x + threadIdx.x;
  if (i < NBTD) {
    size_t bt = i / D_;
    int d = (int)(i % D_);
    int tok = xseq[bt];
    ws[OFF_ER + i] = encw[(size_t)tok * D2_ + d];
    ws[OFF_EI + i] = encw[(size_t)tok * D2_ + D_ + d];
  }
}

__global__ void cvt_bf16_kernel(const float* __restrict__ src, bf16* __restrict__ dst,
                                size_t n) {
  size_t i = (size_t)blockIdx.x * blockDim.x + threadIdx.x;
  if (i < n) dst[i] = (bf16)src[i];
}

// wcat layout: [j][n][k] (k contiguous) so B fragments are b128 loads
__global__ void prep_wcat_kernel(const float* qwr, const float* qbr, const float* qwi, const float* qbi,
                                 const float* kwr, const float* kbr, const float* kwi, const float* kbi,
                                 const float* vwr, const float* vbr, const float* vwi, const float* vbi,
                                 float* ws) {
  bf16* wcat = (bf16*)(ws + OFF_WCAT);
  float* biascat = ws + OFF_BIAS;
  size_t i = (size_t)blockIdx.x * blockDim.x + threadIdx.x;
  constexpr size_t tot = (size_t)6 * D_ * D2_;
  if (i >= tot) return;
  int j = (int)(i / (D_ * D2_));
  int rem = (int)(i % (D_ * D2_));
  int n = rem / D2_, k = rem % D2_;
  int pair = j >> 1, im = j & 1;
  const float* wr = pair == 0 ? qwr : (pair == 1 ? kwr : vwr);
  const float* wi = pair == 0 ? qwi : (pair == 1 ? kwi : vwi);
  float val;
  if (im == 0) val = (k < D_) ? wr[n * D_ + k] : -wi[n * D_ + (k - D_)];
  else         val = (k < D_) ? wi[n * D_ + k] :  wr[n * D_ + (k - D_)];
  wcat[i] = (bf16)val;
  if (k == 0) {
    const float* br = pair == 0 ? qbr : (pair == 1 ? kbr : vbr);
    const float* bi = pair == 0 ? qbi : (pair == 1 ? kbi : vbi);
    biascat[j * D_ + n] = (im == 0) ? (br[n] - bi[n]) : (br[n] + bi[n]);
  }
}

__global__ void embsq_kernel(const float* __restrict__ vq, float* ws) {
  int c = threadIdx.x;
  if (c < C_) {
    float s = 0.f;
    for (int d = 0; d < D2_; ++d) { float v = vq[(size_t)c * D2_ + d]; s += v * v; }
    ws[OFF_EMBSQ + c] = s;
  }
}

// ------------------- per-step kernel 1: LN + VQ + halting -------------------
__global__ void pre_kernel(float* ws,
                           const float* __restrict__ lnrg, const float* __restrict__ lnrb,
                           const float* __restrict__ lnig, const float* __restrict__ lnib,
                           const float* __restrict__ vq_emb,
                           const float* halt_bias, const float* halt_scale, const float* input_gate,
                           int t, int r) {
  __shared__ float sred[256];
  __shared__ float sflat[512];
  __shared__ float dval[128];
  __shared__ int   didx[128];

  int b = blockIdx.x, tid = threadIdx.x;
  float* gwr = ws + OFF_GWR + (size_t)b * D_;
  float* gwi = ws + OFF_GWI + (size_t)b * D_;

  float xr = gwr[tid], xi = gwi[tid];
  if (r == 0) {
    float alpha = sigmoidf_(input_gate[0]);
    xr = alpha * xr + (1.f - alpha) * ws[OFF_ER + ((size_t)b * T_ + t) * D_ + tid];
    xi = alpha * xi + (1.f - alpha) * ws[OFF_EI + ((size_t)b * T_ + t) * D_ + tid];
    gwr[tid] = xr; gwi[tid] = xi;
    ws[OFF_PANG + (size_t)b * D_ + tid] = atan2f(xi, xr);
    if (tid == 0) ws[OFF_ACT + b] = 1.0f;
  }
  __syncthreads();

  float mr = block_reduce_sum(xr, sred) * (1.f / D_);
  float vr = block_reduce_sum((xr - mr) * (xr - mr), sred) * (1.f / D_);
  float csr = (xr - mr) * rsqrtf(vr + 1e-5f) * lnrg[tid] + lnrb[tid];
  float mi = block_reduce_sum(xi, sred) * (1.f / D_);
  float vi = block_reduce_sum((xi - mi) * (xi - mi), sred) * (1.f / D_);
  float csi = (xi - mi) * rsqrtf(vi + 1e-5f) * lnig[tid] + lnib[tid];

  ws[OFF_CSR + (size_t)b * D_ + tid] = csr;
  ws[OFF_CSI + (size_t)b * D_ + tid] = csi;
  float* flat = ws + OFF_FLAT + (size_t)b * D2_;
  flat[tid] = csr; flat[tid + D_] = csi;
  sflat[tid] = csr; sflat[tid + D_] = csi;
  __syncthreads();

  float fs = block_reduce_sum(csr * csr + csi * csi, sred);

  if (tid < C_) {
    const float* code = vq_emb + (size_t)tid * D2_;
    float dot = 0.f;
    for (int d = 0; d < D2_; ++d) dot += sflat[d] * code[d];
    dval[tid] = fs + ws[OFF_EMBSQ + tid] - 2.f * dot;
    didx[tid] = tid;
  }
  __syncthreads();
  for (int s = 64; s > 0; s >>= 1) {
    if (tid < s) {
      if (dval[tid + s] < dval[tid]) { dval[tid] = dval[tid + s]; didx[tid] = didx[tid + s]; }
    }
    __syncthreads();
  }
  int idx = didx[0];
  __syncthreads();

  const float* code = vq_emb + (size_t)idx * D2_;
  float e1 = sflat[tid] - code[tid];
  float e2 = sflat[tid + D_] - code[tid + D_];
  float vql = block_reduce_sum(e1 * e1 + e2 * e2, sred) * (1.25f / (float)D2_);

  float* zq = ws + OFF_ZQ + (size_t)b * D2_;
  zq[tid] = code[tid]; zq[tid + D_] = code[tid + D_];

  if (tid == 0) {
    float hb = log1pf(expf(halt_bias[0]));
    float hs = log1pf(expf(halt_scale[0]));
    float hp = sigmoidf_(hb - hs * vql);
    ws[OFF_STOP + b] = (hp > 0.5f) ? 1.f : 0.f;
    int li = (t * R_ + r) * B_ + b;
    float am = ws[OFF_ACT + b];
    ws[OFF_AMLOG + li] = am;
    ws[OFF_VQLOG + li] = vql;
    ((int*)(ws + OFF_IDXLOG))[li] = idx;
  }
}

// ------------------- per-step kernel 2: q/k/v complex linears via WMMA -------------------
// grid: 12 blocks = j(6) x mt(2); 8 waves x 2 N-tiles each -> N=256
__global__ void clinear_wmma_kernel(const float* __restrict__ flat,
                                    const bf16* __restrict__ wcat,
                                    const float* __restrict__ biascat,
                                    float* __restrict__ qkv) {
  __shared__ alignas(16) bf16 sA[16 * 512];
  int tid = threadIdx.x;
  int bid = blockIdx.x;
  int j  = bid >> 1;
  int mt = bid & 1;
  int m0 = mt * 16;

  for (int i = tid; i < 16 * 512; i += 256)
    sA[i] = (bf16)flat[(size_t)(m0 + (i >> 9)) * D2_ + (i & 511)];
  __syncthreads();

  int w = tid >> 5, lane = tid & 31;
  int n0 = w * 32;
  int l15 = lane & 15, h = lane >> 4;
  const bf16* wj = wcat + (size_t)j * D_ * D2_;
  const bf16* arow  = sA + l15 * 512;
  const bf16* brow0 = wj + (size_t)(n0 + l15) * D2_;
  const bf16* brow1 = wj + (size_t)(n0 + 16 + l15) * D2_;

  v8f acc0 = {0.f, 0.f, 0.f, 0.f, 0.f, 0.f, 0.f, 0.f};
  v8f acc1 = {0.f, 0.f, 0.f, 0.f, 0.f, 0.f, 0.f, 0.f};
#pragma unroll
  for (int kk = 0; kk < D2_; kk += 32) {
    v16bf a  = load_frag(arow + kk, h);
    v16bf b0 = load_frag(brow0 + kk, h);
    v16bf b1 = load_frag(brow1 + kk, h);
    acc0 = __builtin_amdgcn_wmma_f32_16x16x32_bf16(false, a, false, b0, (short)0, acc0,
                                                   false, false);
    acc1 = __builtin_amdgcn_wmma_f32_16x16x32_bf16(false, a, false, b1, (short)0, acc1,
                                                   false, false);
  }
#pragma unroll
  for (int rr = 0; rr < 8; ++rr) {
    int mm = m0 + rr + 8 * h;
    int nn0 = n0 + l15, nn1 = n0 + 16 + l15;
    qkv[(size_t)j * B_ * D_ + (size_t)mm * D_ + nn0] = acc0[rr] + biascat[j * D_ + nn0];
    qkv[(size_t)j * B_ * D_ + (size_t)mm * D_ + nn1] = acc1[rr] + biascat[j * D_ + nn1];
  }
}

// ------------------- per-step kernel 3: attention/memory/arbitration/update -------------------
__global__ void post_kernel(float* ws,
                            const float* __restrict__ gate_w, const float* gate_b,
                            const float* __restrict__ addr_w, const float* __restrict__ addr_b,
                            const float* __restrict__ mlnrg, const float* __restrict__ mlnrb,
                            const float* __restrict__ mlnig, const float* __restrict__ mlnib,
                            const float* __restrict__ arb_w, const float* __restrict__ arb_b,
                            int t, int r) {
  __shared__ float sred[256];
  __shared__ float sflat[512];
  __shared__ float sim[32], attn[32], wws[32], eff[32];
  __shared__ float g3[3];
  __shared__ float redm_r[256], redv_r[256], redm_i[256], redv_i[256];
  __shared__ float meanr[32], varr[32], meani[32], vari[32];

  int b = blockIdx.x, tid = threadIdx.x;
  float am = ws[OFF_ACT + b];
  bool keep = am > 0.5f;

  const float* flat = ws + OFF_FLAT + (size_t)b * D2_;
  sflat[tid] = flat[tid]; sflat[tid + D_] = flat[tid + D_];
  __syncthreads();
  float csr = sflat[tid], csi = sflat[tid + D_];

  const float* qkv = ws + OFF_QKV;
  size_t qo = (size_t)b * D_ + tid;
  float qr = qkv[0 * B_ * D_ + qo], qi = qkv[1 * B_ * D_ + qo];
  float kr = qkv[2 * B_ * D_ + qo], ki = qkv[3 * B_ * D_ + qo];
  float hr = qkv[4 * B_ * D_ + qo], hi = qkv[5 * B_ * D_ + qo];

  float gsum = block_reduce_sum(qr * kr + qi * ki, sred);
  float gate = sigmoidf_(gsum);
  float g_r = hr * gate, g_i = hi * gate;

  float wgdot = block_reduce_sum(csr * gate_w[tid] + csi * gate_w[tid + D_], sred);

  float* memr = ws + OFF_MEMR + (size_t)b * S_ * D_;
  float* memi = ws + OFF_MEMI + (size_t)b * S_ * D_;
  if (tid < S_) {
    float acc = 0.f;
    for (int d = 0; d < D_; ++d)
      acc += memr[tid * D_ + d] * sflat[d] + memi[tid * D_ + d] * sflat[D_ + d];
    sim[tid] = acc;
    const float* aw = addr_w + (size_t)tid * D2_;
    float a2 = 0.f;
    for (int d = 0; d < D2_; ++d) a2 += sflat[d] * aw[d];
    wws[tid] = a2 + addr_b[tid];
  }
  if (tid < 3) {
    const float* aw = arb_w + (size_t)tid * D2_;
    float a3 = 0.f;
    for (int d = 0; d < D2_; ++d) a3 += sflat[d] * aw[d];
    g3[tid] = a3 + arb_b[tid];
  }
  __syncthreads();

  if (tid == 0) {
    float mx = -1e30f;
    for (int s = 0; s < S_; ++s) mx = fmaxf(mx, sim[s]);
    float sm = 0.f;
    for (int s = 0; s < S_; ++s) { attn[s] = expf(sim[s] - mx); sm += attn[s]; }
    for (int s = 0; s < S_; ++s) attn[s] /= sm;
    float wg = sigmoidf_(wgdot + gate_b[0]);
    float mx2 = -1e30f;
    for (int s = 0; s < S_; ++s) mx2 = fmaxf(mx2, wws[s]);
    float s2 = 0.f;
    for (int s = 0; s < S_; ++s) { wws[s] = expf(wws[s] - mx2); s2 += wws[s]; }
    for (int s = 0; s < S_; ++s) eff[s] = wg * wws[s] / s2;
    float gm = fmaxf(fmaxf(g3[0], g3[1]), g3[2]);
    float e0 = expf(g3[0] - gm), e1 = expf(g3[1] - gm), e2 = expf(g3[2] - gm);
    float es = e0 + e1 + e2;
    g3[0] = e0 / es; g3[1] = e1 / es; g3[2] = e2 / es;
  }
  __syncthreads();

  float m_r = 0.f, m_i = 0.f;
  for (int s = 0; s < S_; ++s) {
    m_r += attn[s] * memr[s * D_ + tid];
    m_i += attn[s] * memi[s * D_ + tid];
  }

  const float* zq = ws + OFF_ZQ + (size_t)b * D2_;
  float u_r = g3[0] * g_r + g3[1] * m_r + g3[2] * zq[tid];
  float u_i = g3[0] * g_i + g3[1] * m_i + g3[2] * zq[tid + D_];
  float cand_r = 0.6f * csr + 0.4f * u_r;
  float cand_i = 0.6f * csi + 0.4f * u_i;
  float ang = atan2f(cand_i, cand_r);
  float* pang = ws + OFF_PANG + (size_t)b * D_;
  float diff = fabsf(ang - pang[tid]);
  diff = fminf(diff, TWO_PI_ - diff);
  float dm = block_reduce_sum(diff, sred) * (1.f / D_);
  if (tid == 0) ws[OFF_PHLOG + (size_t)(t * R_ + r) * B_ + b] = am * dm;

  int sg = tid >> 3, jl = tid & 7;
  float s0r = 0.f, s1r = 0.f, s0i = 0.f, s1i = 0.f;
  for (int d = jl * 32; d < jl * 32 + 32; ++d) {
    float vvr = (1.f - eff[sg]) * memr[sg * D_ + d] + eff[sg] * sflat[d];
    float vvi = (1.f - eff[sg]) * memi[sg * D_ + d] + eff[sg] * sflat[D_ + d];
    s0r += vvr; s1r += vvr * vvr; s0i += vvi; s1i += vvi * vvi;
  }
  redm_r[tid] = s0r; redv_r[tid] = s1r; redm_i[tid] = s0i; redv_i[tid] = s1i;
  __syncthreads();
  if (jl == 0) {
    float ar = 0.f, a2r = 0.f, ai = 0.f, a2i = 0.f;
    for (int q = 0; q < 8; ++q) {
      ar += redm_r[sg * 8 + q]; a2r += redv_r[sg * 8 + q];
      ai += redm_i[sg * 8 + q]; a2i += redv_i[sg * 8 + q];
    }
    float mmr = ar / D_, mmi = ai / D_;
    meanr[sg] = mmr; varr[sg] = a2r / D_ - mmr * mmr;
    meani[sg] = mmi; vari[sg] = a2i / D_ - mmi * mmi;
  }
  __syncthreads();
  if (keep) {
    float irr = rsqrtf(varr[sg] + 1e-5f), iri = rsqrtf(vari[sg] + 1e-5f);
    for (int d = jl * 32; d < jl * 32 + 32; ++d) {
      float vvr = (1.f - eff[sg]) * memr[sg * D_ + d] + eff[sg] * sflat[d];
      float vvi = (1.f - eff[sg]) * memi[sg * D_ + d] + eff[sg] * sflat[D_ + d];
      memr[sg * D_ + d] = (vvr - meanr[sg]) * irr * mlnrg[d] + mlnrb[d];
      memi[sg * D_ + d] = (vvi - meani[sg]) * iri * mlnig[d] + mlnib[d];
    }
  }

  float* gwr = ws + OFF_GWR + (size_t)b * D_;
  float* gwi = ws + OFF_GWI + (size_t)b * D_;
  float ngr = keep ? cand_r : gwr[tid];
  float ngi = keep ? cand_i : gwi[tid];
  gwr[tid] = ngr; gwi[tid] = ngi;
  if (keep) pang[tid] = ang;
  if (tid == 0) ws[OFF_ACT + b] = am * (1.f - ws[OFF_STOP + b]);
  if (r == R_ - 1) {
    float* os = ws + OFF_OUTSEQ + ((size_t)b * T_ + t) * D2_;
    os[tid] = ngr; os[tid + D_] = ngi;
  }
}

// ------------------- finalize: stats + fin_idx -------------------
__global__ void stats_kernel(float* ws, float* out) {
  __shared__ float part[32][5];
  int tid = threadIdx.x;
  if (tid < T_) {
    int t = tid;
    float vq_acc[B_], tsb[B_], ph[B_];
    for (int b = 0; b < B_; ++b) { vq_acc[b] = 0.f; tsb[b] = 0.f; ph[b] = 0.f; }
    float tot_ent = 0.f; int steps = 0;
    for (int r = 0; r < R_; ++r) {
      int base = (t * R_ + r) * B_;
      float asum = 0.f;
      int counts[C_];
      for (int c = 0; c < C_; ++c) counts[c] = 0;
      for (int b = 0; b < B_; ++b) {
        float am = ws[OFF_AMLOG + base + b];
        asum += am;
        vq_acc[b] += am * ws[OFF_VQLOG + base + b];
        tsb[b] += am;
        ph[b] += ws[OFF_PHLOG + base + b];
        counts[((int*)(ws + OFF_IDXLOG))[base + b]]++;
      }
      if (asum > 0.f) {
        float ent = 0.f;
        for (int c = 0; c < C_; ++c) {
          float p = counts[c] / (float)B_;
          ent -= p * logf(p + 1e-10f);
        }
        tot_ent += ent; steps++;
      }
    }
    float svq = 0.f, spond = 0.f, sph = 0.f, sdep = 0.f;
    for (int b = 0; b < B_; ++b) {
      svq += vq_acc[b] / fmaxf(tsb[b], 1.f);
      spond += PONDER_ * tsb[b];
      sph += ph[b];
      sdep += tsb[b];
    }
    part[t][0] = svq / B_;
    part[t][1] = spond / B_;
    part[t][2] = sph / B_;
    part[t][3] = tot_ent / fmaxf((float)steps, 1.f);
    part[t][4] = sdep / B_;
    for (int b = 0; b < B_; ++b) {
      int fin = 0;
      for (int r = 0; r < R_; ++r) {
        int base = (t * R_ + r) * B_ + b;
        if (ws[OFF_AMLOG + base] > 0.5f) fin = ((int*)(ws + OFF_IDXLOG))[base];
      }
      ((int*)(out + OUT_IDX))[b * T_ + t] = fin;
    }
  }
  __syncthreads();
  if (tid == 0) {
    float s[5] = {0.f, 0.f, 0.f, 0.f, 0.f};
    for (int t = 0; t < T_; ++t)
      for (int q = 0; q < 5; ++q) s[q] += part[t][q];
    for (int q = 0; q < 5; ++q) out[OUT_STATS + q] = s[q] / (float)T_;
  }
}

__global__ void copy_mem_kernel(const float* __restrict__ ws, float* __restrict__ out) {
  size_t i = (size_t)blockIdx.x * blockDim.x + threadIdx.x;
  if (i < (size_t)B_ * S_ * D_) {
    out[OUT_MEMR + i] = ws[OFF_MEMR + i];
    out[OUT_MEMI + i] = ws[OFF_MEMI + i];
  }
}

// ------------------- decoder GEMM: (1024x512) @ (512x32000) via bf16 WMMA -------------------
// grid: (125, 16); block = 256 threads = 8 waves.
// 64 M-rows staged in LDS per block (64KB); each wave computes a 64x32 patch:
// 4 M-tiles x 2 N-tiles = 8 accumulators, so each B fragment feeds 4 WMMAs
// (4x less L2 traffic on dec_w than a 16-row block).
__global__ void decoder_wmma_kernel(const bf16* __restrict__ outbf,
                                    const bf16* __restrict__ decbf,
                                    const float* __restrict__ dec_b,
                                    float* __restrict__ logits) {
  __shared__ alignas(16) bf16 sA[64 * 512];
  int tid = threadIdx.x;
  int m0 = blockIdx.y * 64;

  const v8bf* srcv = (const v8bf*)(outbf + (size_t)m0 * D2_);
  v8bf* dstv = (v8bf*)sA;
  for (int i = tid; i < 64 * 512 / 8; i += 256) dstv[i] = srcv[i];
  __syncthreads();

  int w = tid >> 5, lane = tid & 31;
  int n0 = blockIdx.x * 256 + w * 32;
  int l15 = lane & 15, h = lane >> 4;

  const bf16* brow0 = decbf + (size_t)(n0 + l15) * D2_;
  const bf16* brow1 = decbf + (size_t)(n0 + 16 + l15) * D2_;

  v8f acc[4][2];
#pragma unroll
  for (int mt = 0; mt < 4; ++mt) {
    acc[mt][0] = (v8f){0.f, 0.f, 0.f, 0.f, 0.f, 0.f, 0.f, 0.f};
    acc[mt][1] = (v8f){0.f, 0.f, 0.f, 0.f, 0.f, 0.f, 0.f, 0.f};
  }

#pragma unroll 4
  for (int kk = 0; kk < D2_; kk += 32) {
    if (kk + 128 < D2_) {
      __builtin_prefetch(brow0 + kk + 128, 0, 1);
      __builtin_prefetch(brow1 + kk + 128, 0, 1);
    }
    v16bf b0 = load_frag(brow0 + kk, h);
    v16bf b1 = load_frag(brow1 + kk, h);
#pragma unroll
    for (int mt = 0; mt < 4; ++mt) {
      v16bf a = load_frag(sA + (size_t)(mt * 16 + l15) * 512 + kk, h);
      acc[mt][0] = __builtin_amdgcn_wmma_f32_16x16x32_bf16(false, a, false, b0, (short)0,
                                                           acc[mt][0], false, false);
      acc[mt][1] = __builtin_amdgcn_wmma_f32_16x16x32_bf16(false, a, false, b1, (short)0,
                                                           acc[mt][1], false, false);
    }
  }

#pragma unroll
  for (int mt = 0; mt < 4; ++mt) {
#pragma unroll
    for (int rr = 0; rr < 8; ++rr) {
      int mm = m0 + mt * 16 + rr + 8 * h;
      int nn0 = n0 + l15, nn1 = n0 + 16 + l15;
      logits[(size_t)mm * V_ + nn0] = acc[mt][0][rr] + dec_b[nn0];
      logits[(size_t)mm * V_ + nn1] = acc[mt][1][rr] + dec_b[nn1];
    }
  }
}

// ------------------- launch -------------------
extern "C" void kernel_launch(void* const* d_in, const int* in_sizes, int n_in,
                              void* d_out, int out_size, void* d_ws, size_t ws_size,
                              hipStream_t stream) {
  const int*   x_seq  = (const int*)d_in[0];
  const float* enc_w  = (const float*)d_in[1];
  const float* lnrg   = (const float*)d_in[2];
  const float* lnrb   = (const float*)d_in[3];
  const float* lnig   = (const float*)d_in[4];
  const float* lnib   = (const float*)d_in[5];
  const float* qwr    = (const float*)d_in[6];
  const float* qbr    = (const float*)d_in[7];
  const float* qwi    = (const float*)d_in[8];
  const float* qbi    = (const float*)d_in[9];
  const float* kwr    = (const float*)d_in[10];
  const float* kbr    = (const float*)d_in[11];
  const float* kwi    = (const float*)d_in[12];
  const float* kbi    = (const float*)d_in[13];
  const float* vwr    = (const float*)d_in[14];
  const float* vbr    = (const float*)d_in[15];
  const float* vwi    = (const float*)d_in[16];
  const float* vbi    = (const float*)d_in[17];
  const float* gate_w = (const float*)d_in[18];
  const float* gate_b = (const float*)d_in[19];
  const float* addr_w = (const float*)d_in[20];
  const float* addr_b = (const float*)d_in[21];
  const float* mlnrg  = (const float*)d_in[22];
  const float* mlnrb  = (const float*)d_in[23];
  const float* mlnig  = (const float*)d_in[24];
  const float* mlnib  = (const float*)d_in[25];
  const float* vq_emb = (const float*)d_in[26];
  const float* arb_w  = (const float*)d_in[27];
  const float* arb_b  = (const float*)d_in[28];
  const float* dec_w  = (const float*)d_in[29];
  const float* dec_b  = (const float*)d_in[30];
  const float* hbias  = (const float*)d_in[31];
  const float* hscale = (const float*)d_in[32];
  const float* igate  = (const float*)d_in[33];

  float* ws  = (float*)d_ws;
  float* out = (float*)d_out;
  bf16* wcat   = (bf16*)(ws + OFF_WCAT);
  bf16* decbf  = (bf16*)(ws + OFF_DECWBF);
  bf16* outbf  = (bf16*)(ws + OFF_OUTBF);

  zero_state_kernel<<<(unsigned)((ZERO_N + 255) / 256), 256, 0, stream>>>(ws);
  embed_kernel<<<(unsigned)((NBTD + 255) / 256), 256, 0, stream>>>(x_seq, enc_w, ws);
  prep_wcat_kernel<<<(unsigned)(((size_t)6 * D_ * D2_ + 255) / 256), 256, 0, stream>>>(
      qwr, qbr, qwi, qbi, kwr, kbr, kwi, kbi, vwr, vbr, vwi, vbi, ws);
  embsq_kernel<<<1, 128, 0, stream>>>(vq_emb, ws);
  {
    size_t n = (size_t)V_ * D2_;
    cvt_bf16_kernel<<<(unsigned)((n + 255) / 256), 256, 0, stream>>>(dec_w, decbf, n);
  }

  for (int t = 0; t < T_; ++t) {
    for (int r = 0; r < R_; ++r) {
      pre_kernel<<<B_, 256, 0, stream>>>(ws, lnrg, lnrb, lnig, lnib, vq_emb,
                                         hbias, hscale, igate, t, r);
      clinear_wmma_kernel<<<12, 256, 0, stream>>>(ws + OFF_FLAT, wcat,
                                                  ws + OFF_BIAS, ws + OFF_QKV);
      post_kernel<<<B_, 256, 0, stream>>>(ws, gate_w, gate_b, addr_w, addr_b,
                                          mlnrg, mlnrb, mlnig, mlnib, arb_w, arb_b, t, r);
    }
  }

  stats_kernel<<<1, 256, 0, stream>>>(ws, out);
  copy_mem_kernel<<<(unsigned)(((size_t)B_ * S_ * D_ + 255) / 256), 256, 0, stream>>>(ws, out);
  {
    size_t n = (size_t)B_ * T_ * D2_;
    cvt_bf16_kernel<<<(unsigned)((n + 255) / 256), 256, 0, stream>>>(ws + OFF_OUTSEQ, outbf, n);
  }
  decoder_wmma_kernel<<<dim3(125, 16), 256, 0, stream>>>(outbf, decbf, dec_b,
                                                         out + OUT_LOGITS);
}
